// Single_head_attention_6356551598393
// MI455X (gfx1250) — matrix operand order repaired and checked by reading, not compiled, not run on hardware
//
#include <hip/hip_runtime.h>
#include <stdint.h>

// ---------------------------------------------------------------------------
// Single-head attention for MI455X (gfx1250, wave32, WMMA + TDM).
//   q = X_q Wq + bq ; k = X_k Wk + bk ; v = X_v Wv + bv   (bf16 out, f32 acc)
//   O = softmax(q k^T / sqrt(512)) v                       (flash, bf16 WMMA)
// K/V/Q tiles are staged into LDS by the Tensor Data Mover (double buffered).
// ---------------------------------------------------------------------------

#define HIDDEN 1024
#define DHEAD  512
#define SEQ    2048
#define NBATCH 8

#define BR 64      // q rows per attention block (4 waves x 16)
#define BC 32      // kv columns per flash iteration
#define AWAVES 4
#define NTILES (SEQ / BC)

typedef __attribute__((ext_vector_type(16))) __bf16 v16bf;
typedef __attribute__((ext_vector_type(8)))  float  v8f;
typedef uint32_t v4u __attribute__((ext_vector_type(4)));
typedef int32_t  v8i __attribute__((ext_vector_type(8)));
typedef int32_t  v4i __attribute__((ext_vector_type(4)));

union Frag {           // one WMMA 16-bit operand fragment = 8 VGPRs / lane
    v16bf v;
    uint4 q[2];
};

// round-to-nearest-even f32 -> bf16 (bit ops only)
__device__ __forceinline__ uint32_t pack2_bf16(float a, float b) {
    uint32_t ua = __float_as_uint(a), ub = __float_as_uint(b);
    ua = (ua + 0x7FFFu + ((ua >> 16) & 1u)) >> 16;
    ub = (ub + 0x7FFFu + ((ub >> 16) & 1u)) >> 16;
    return ua | (ub << 16);
}
__device__ __forceinline__ uint16_t f2bf(float a) {
    uint32_t ua = __float_as_uint(a);
    return (uint16_t)((ua + 0x7FFFu + ((ua >> 16) & 1u)) >> 16);
}

// ---------------------------------------------------------------------------
// TDM: issue a 2D tile load (bf16 elements) global -> LDS via TENSOR_LOAD_TO_LDS.
// Descriptor bitfields per CDNA5 ISA 8.3/8.4. All operands are wave-uniform.
//   tensor_d0/d1 : tensor extent (elements) in dim0 (contiguous) / dim1 (rows)
//   tile_d0/d1   : tile extent  (elements); tile rows packed contiguously in LDS
//   stride0      : row stride of the tensor in elements
// ---------------------------------------------------------------------------
__device__ __forceinline__ void tdm_load_2d(uint32_t lds_off, const uint16_t* gptr,
                                            uint32_t tensor_d0, uint32_t tensor_d1,
                                            uint32_t tile_d0, uint32_t tile_d1,
                                            uint32_t stride0)
{
    uint64_t ga = (uint64_t)(uintptr_t)gptr;
    v4u g0;
    g0[0] = 1u;                                   // count=1 (valid), user mode
    g0[1] = lds_off;                              // LDS byte address
    g0[2] = (uint32_t)ga;                         // global_addr[31:0]
    g0[3] = (uint32_t)(ga >> 32) | (2u << 30);    // global_addr[56:32] | type=2

    v8i g1;
    g1[0] = (int)(1u << 16);                                  // data_size=1 (2B); wg_mask=0
    g1[1] = (int)((tensor_d0 & 0xFFFFu) << 16);               // tensor_dim0[15:0] @ bit48
    g1[2] = (int)((tensor_d0 >> 16) | ((tensor_d1 & 0xFFFFu) << 16)); // td0 hi | td1 lo
    g1[3] = (int)((tensor_d1 >> 16) | (tile_d0 << 16));       // td1 hi | tile_dim0
    g1[4] = (int)tile_d1;                                     // tile_dim1 | tile_dim2=0
    g1[5] = (int)stride0;                                     // tensor_dim0_stride[31:0]
    g1[6] = 0;                                                // stride0 hi | stride1 lo
    g1[7] = 0;                                                // stride1 hi

    v4i g2 = {0, 0, 0, 0};
    v4i g3 = {0, 0, 0, 0};
#if defined(__clang_major__) && (__clang_major__ >= 23)
    v8i g4 = {0, 0, 0, 0, 0, 0, 0, 0};
    __builtin_amdgcn_tensor_load_to_lds(g0, g1, g2, g3, g4, 0);
#else
    __builtin_amdgcn_tensor_load_to_lds(g0, g1, g2, g3, 0);
#endif
}

__device__ __forceinline__ uint32_t lds_off_of(const void* p) {
    return (uint32_t)(uintptr_t)p;   // LDS aperture: low 32 bits = LDS byte offset
}

// ---------------------------------------------------------------------------
// Projection GEMM: out[16384 x 512](bf16) = X[16384 x 1024](f32) * W[1024 x 512](f32) + bias
// Block: 256 thr (8 waves), tile 128 rows x 64 cols; wave w -> rows [16w,16w+16),
// 4 col sub-tiles of 16. K staged 32 wide per iteration (f32 -> bf16 in flight,
// so VALU staging, not TDM). transposeOut=1 stores per-batch [b][d][s] (for V).
// ---------------------------------------------------------------------------
__global__ __launch_bounds__(256) void proj_kernel(
    const float* __restrict__ X, const float* __restrict__ W,
    const float* __restrict__ bias, uint16_t* __restrict__ out,
    int transposeOut)
{
    __shared__ uint16_t ldsA[128 * 32];   // [row][k] bf16, row = 64 B
    __shared__ uint16_t ldsW[64 * 32];    // [col][k] bf16 (W transposed), row = 64 B

    const int tid  = threadIdx.x;
    const int lane = tid & 31;
    const int wave = tid >> 5;
    const int hh   = lane >> 4;           // lane half (ISA fragment layouts)
    const int ln   = lane & 15;
    const int row0 = blockIdx.y * 128;
    const int col0 = blockIdx.x * 64;

    v8f acc[4];
#pragma unroll
    for (int t = 0; t < 4; ++t)
#pragma unroll
        for (int i = 0; i < 8; ++i) acc[t][i] = 0.0f;

    for (int kb = 0; kb < HIDDEN; kb += 32) {
        // ---- stage A tile 128x32 f32 -> bf16 (each thread: 16 elems) ----
        {
            int r  = tid >> 1;
            int kh = (tid & 1) * 16;
            const float4* src =
                reinterpret_cast<const float4*>(X + (size_t)(row0 + r) * HIDDEN + kb + kh);
            float4 f0 = src[0], f1 = src[1], f2 = src[2], f3 = src[3];
            uint4 p0, p1;
            p0.x = pack2_bf16(f0.x, f0.y); p0.y = pack2_bf16(f0.z, f0.w);
            p0.z = pack2_bf16(f1.x, f1.y); p0.w = pack2_bf16(f1.z, f1.w);
            p1.x = pack2_bf16(f2.x, f2.y); p1.y = pack2_bf16(f2.z, f2.w);
            p1.z = pack2_bf16(f3.x, f3.y); p1.w = pack2_bf16(f3.z, f3.w);
            uint4* dst = reinterpret_cast<uint4*>(ldsA + r * 32 + kh);
            dst[0] = p0; dst[1] = p1;
        }
        // ---- stage W tile 32x64 f32 -> LDS transposed [col][k] bf16 ----
        {
            int n  = tid >> 2;
            int kq = (tid & 3) * 8;
            const float* src = W + (size_t)(kb + kq) * DHEAD + (col0 + n);
            float f[8];
#pragma unroll
            for (int j = 0; j < 8; ++j) f[j] = src[(size_t)j * DHEAD];
            uint4 p;
            p.x = pack2_bf16(f[0], f[1]); p.y = pack2_bf16(f[2], f[3]);
            p.z = pack2_bf16(f[4], f[5]); p.w = pack2_bf16(f[6], f[7]);
            *reinterpret_cast<uint4*>(ldsW + n * 32 + kq) = p;
        }
        __syncthreads();

        // ---- WMMA: A frag (16x32 bf16, ISA layout), 4 B frags ----
        const uint4* ap =
            reinterpret_cast<const uint4*>(ldsA + (wave * 16 + ln) * 32);
        Frag a;
        a.q[0] = ap[hh];        // K {0,2,4,6}+8h pairs
        a.q[1] = ap[2 + hh];    // K {16,..,22}+8h pairs
#pragma unroll
        for (int ct = 0; ct < 4; ++ct) {
            const uint4* bp =
                reinterpret_cast<const uint4*>(ldsW + (16 * ct + ln) * 32);
            Frag b;
            b.q[0] = bp[hh];          // K = 8h .. 8h+7 pairs
            b.q[1] = bp[2 + hh];      // K = 16+8h .. 16+8h+7 pairs
            acc[ct] = __builtin_amdgcn_wmma_f32_16x16x32_bf16(
                false, a.v, false, b.v, (short)0, acc[ct], false, false);
        }
        __syncthreads();
    }

    // ---- epilogue: + bias, f32 -> bf16, store (optionally transposed) ----
#pragma unroll
    for (int ct = 0; ct < 4; ++ct) {
        int col = col0 + 16 * ct + ln;
        float bv = bias[col];
        int rbase = row0 + wave * 16 + 8 * hh;
#pragma unroll
        for (int i = 0; i < 8; ++i) {
            float val = acc[ct][i] + bv;
            int r = rbase + i;
            size_t idx;
            if (transposeOut) {
                int bb = r >> 11, s = r & 2047;      // batch, seq (2048 rows/batch)
                idx = (size_t)bb * (DHEAD * SEQ) + (size_t)col * SEQ + s;
            } else {
                idx = (size_t)r * DHEAD + col;
            }
            out[idx] = f2bf(val);
        }
    }
}

// ---------------------------------------------------------------------------
// Flash attention: block = 4 waves / 64 q rows; wave owns 16 rows, full D=512.
// LDS: Q[64][512] resident + double-buffered K[32][512], VT[512][32] staged by
// the Tensor Data Mover + P scratch[4][16][32].
// ---------------------------------------------------------------------------
extern __shared__ __align__(16) uint16_t smem_dyn[];

__global__ __launch_bounds__(128) void attn_kernel(
    const uint16_t* __restrict__ qg,   // [b][s][d] bf16
    const uint16_t* __restrict__ kg,   // [b][s][d] bf16
    const uint16_t* __restrict__ vtg,  // [b][d][s] bf16 (transposed)
    float* __restrict__ out)           // [b][s][d] f32
{
    uint16_t* Qs    = smem_dyn;                       // 64*512
    uint16_t* Kb[2] = { Qs + BR * DHEAD,              // 32*512 each
                        Qs + BR * DHEAD + BC * DHEAD };
    uint16_t* Vb[2] = { Kb[1] + BC * DHEAD,           // 512*32 each (VT: [d][kcol])
                        Kb[1] + BC * DHEAD + DHEAD * BC };
    uint16_t* Ps    = Vb[1] + DHEAD * BC;             // 4*16*32

    const int tid  = threadIdx.x;
    const int lane = tid & 31;
    const int wave = tid >> 5;
    const int hh   = lane >> 4;
    const int ln   = lane & 15;
    const int b    = blockIdx.y;
    const int q0   = blockIdx.x * BR;

    const uint16_t* kbase = kg  + (size_t)b * SEQ * DHEAD;
    const uint16_t* vbase = vtg + (size_t)b * DHEAD * SEQ;

    // ---- TDM prologue: Q tile + first K/V pair (3 outstanding tensor ops) ----
    if (wave == 0) {
        tdm_load_2d(lds_off_of(Qs), qg + ((size_t)b * SEQ + q0) * DHEAD,
                    DHEAD, SEQ, DHEAD, BR, DHEAD);
        tdm_load_2d(lds_off_of(Kb[0]), kbase, DHEAD, SEQ, DHEAD, BC, DHEAD);
        tdm_load_2d(lds_off_of(Vb[0]), vbase, SEQ, DHEAD, BC, DHEAD, SEQ);
    }

    v8f O[32];                         // 16 rows x 512 d accumulator
#pragma unroll
    for (int nt = 0; nt < 32; ++nt)
#pragma unroll
        for (int i = 0; i < 8; ++i) O[nt][i] = 0.0f;
    float m[8], l[8];
#pragma unroll
    for (int i = 0; i < 8; ++i) { m[i] = -1e30f; l[i] = 0.0f; }

    const float scale = 0.04419417382415922f;  // 1/sqrt(512)

    for (int it = 0; it < NTILES; ++it) {
        const int cur = it & 1;
        // issue next tile into the other buffer, then wait for current tile.
        // TDM ops complete in order per wave: TENSORcnt<=2 => first 3/2 done.
        if (wave == 0) {
            if (it + 1 < NTILES) {
                const int nxt = cur ^ 1;
                tdm_load_2d(lds_off_of(Kb[nxt]),
                            kbase + (size_t)(it + 1) * BC * DHEAD,
                            DHEAD, SEQ, DHEAD, BC, DHEAD);
                tdm_load_2d(lds_off_of(Vb[nxt]),
                            vbase + (size_t)(it + 1) * BC,
                            SEQ, DHEAD, BC, DHEAD, SEQ);
                __builtin_amdgcn_s_wait_tensorcnt(2);
            } else {
                __builtin_amdgcn_s_wait_tensorcnt(0);
            }
        }
        __syncthreads();   // current-tile data visible to all waves

        const uint16_t* Ks = Kb[cur];
        const uint16_t* Vs = Vb[cur];
        const uint4* qrow =
            reinterpret_cast<const uint4*>(Qs + (wave * 16 + ln) * DHEAD);

        // ---- scores S = Q K^T : 16 rows x 32 cols (2 col sub-tiles) ----
        v8f st[2];
#pragma unroll
        for (int t = 0; t < 2; ++t)
#pragma unroll
            for (int i = 0; i < 8; ++i) st[t][i] = 0.0f;

#pragma unroll
        for (int ks = 0; ks < DHEAD / 32; ++ks) {
            Frag a;
            a.q[0] = qrow[ks * 4 + hh];
            a.q[1] = qrow[ks * 4 + 2 + hh];
#pragma unroll
            for (int ct = 0; ct < 2; ++ct) {
                const uint4* kp =
                    reinterpret_cast<const uint4*>(Ks + (16 * ct + ln) * DHEAD) + ks * 4;
                Frag bb;
                bb.q[0] = kp[hh];
                bb.q[1] = kp[2 + hh];
                st[ct] = __builtin_amdgcn_wmma_f32_16x16x32_bf16(
                    false, a.v, false, bb.v, (short)0, st[ct], false, false);
            }
        }

        // ---- online softmax (per row: VGPR i <-> row i+8h; cols across lanes) ----
        float alpha[8];
#pragma unroll
        for (int i = 0; i < 8; ++i) {
            float v0 = st[0][i] * scale;
            float v1 = st[1][i] * scale;
            float mx = fmaxf(v0, v1);
            mx = fmaxf(mx, __shfl_xor(mx, 1, 16));
            mx = fmaxf(mx, __shfl_xor(mx, 2, 16));
            mx = fmaxf(mx, __shfl_xor(mx, 4, 16));
            mx = fmaxf(mx, __shfl_xor(mx, 8, 16));
            float mn = fmaxf(m[i], mx);
            float al = __expf(m[i] - mn);
            float p0 = __expf(v0 - mn);
            float p1 = __expf(v1 - mn);
            st[0][i] = p0; st[1][i] = p1;
            float rs = p0 + p1;
            rs += __shfl_xor(rs, 1, 16);
            rs += __shfl_xor(rs, 2, 16);
            rs += __shfl_xor(rs, 4, 16);
            rs += __shfl_xor(rs, 8, 16);
            m[i] = mn;
            l[i] = l[i] * al + rs;
            alpha[i] = al;
        }
        // rescale running output
#pragma unroll
        for (int nt = 0; nt < 32; ++nt)
#pragma unroll
            for (int i = 0; i < 8; ++i) O[nt][i] *= alpha[i];

        // ---- P (C-layout) -> bf16 A-fragment via wave-private LDS scratch ----
        uint16_t* pw = Ps + wave * (16 * 32);
#pragma unroll
        for (int ct = 0; ct < 2; ++ct)
#pragma unroll
            for (int i = 0; i < 8; ++i)
                pw[(i + 8 * hh) * 32 + 16 * ct + ln] = f2bf(st[ct][i]);
        const uint4* pr = reinterpret_cast<const uint4*>(pw + ln * 32);
        Frag pa;
        pa.q[0] = pr[hh];
        pa.q[1] = pr[2 + hh];

        // ---- O += P V : 32 d sub-tiles ----
#pragma unroll
        for (int nt = 0; nt < 32; ++nt) {
            const uint4* vp =
                reinterpret_cast<const uint4*>(Vs + (16 * nt + ln) * BC);
            Frag vf;
            vf.q[0] = vp[hh];
            vf.q[1] = vp[2 + hh];
            O[nt] = __builtin_amdgcn_wmma_f32_16x16x32_bf16(
                false, pa.v, false, vf.v, (short)0, O[nt], false, false);
        }
        __syncthreads();   // all reads of buf[cur] done before it is re-filled
    }

    // ---- normalize and write f32 output ----
#pragma unroll
    for (int i = 0; i < 8; ++i) l[i] = 1.0f / l[i];
#pragma unroll
    for (int nt = 0; nt < 32; ++nt) {
        int c = 16 * nt + ln;
#pragma unroll
        for (int i = 0; i < 8; ++i) {
            int r = q0 + wave * 16 + 8 * hh + i;
            out[((size_t)b * SEQ + r) * DHEAD + c] = O[nt][i] * l[i];
        }
    }
}

// ---------------------------------------------------------------------------
extern "C" void kernel_launch(void* const* d_in, const int* in_sizes, int n_in,
                              void* d_out, int out_size, void* d_ws, size_t ws_size,
                              hipStream_t stream) {
    const float* query = (const float*)d_in[0];
    const float* key_  = (const float*)d_in[1];
    const float* value = (const float*)d_in[2];
    const float* Wq = (const float*)d_in[3];
    const float* bq = (const float*)d_in[4];
    const float* Wk = (const float*)d_in[5];
    const float* bk = (const float*)d_in[6];
    const float* Wv = (const float*)d_in[7];
    const float* bv = (const float*)d_in[8];

    const size_t projElems = (size_t)NBATCH * SEQ * DHEAD;  // 8M elems (16 MB bf16)
    uint16_t* q_ws = (uint16_t*)d_ws;
    uint16_t* k_ws = q_ws + projElems;
    uint16_t* v_ws = k_ws + projElems;   // needs 48 MB of d_ws

    dim3 pgrid(DHEAD / 64, (NBATCH * SEQ) / 128);   // (8, 128)
    dim3 pblk(256);
    proj_kernel<<<pgrid, pblk, 0, stream>>>(query, Wq, bq, q_ws, 0);
    proj_kernel<<<pgrid, pblk, 0, stream>>>(key_,  Wk, bk, k_ws, 0);
    proj_kernel<<<pgrid, pblk, 0, stream>>>(value, Wv, bv, v_ws, 1);  // [b][d][s]

    dim3 agrid(SEQ / BR, NBATCH);                   // (32, 8)
    dim3 ablk(32 * AWAVES);                         // 128 threads = 4 waves
    size_t ldsBytes =
        (size_t)(BR * DHEAD + 2 * BC * DHEAD + 2 * DHEAD * BC + AWAVES * 16 * 32)
        * sizeof(uint16_t);                          // ~197 KB < 320 KB/WGP
    attn_kernel<<<agrid, ablk, ldsBytes, stream>>>(q_ws, k_ws, v_ws, (float*)d_out);
}
